// DeformableAttention_57690000720513
// MI455X (gfx1250) — matrix operand order repaired and checked by reading, not compiled, not run on hardware
//
#include <hip/hip_runtime.h>
#include <hip/hip_bf16.h>
#include <stdint.h>

typedef __attribute__((ext_vector_type(16))) _Float16 f16x16;
typedef __attribute__((ext_vector_type(8)))  _Float16 f16x8;
typedef __attribute__((ext_vector_type(8)))  float    f32x8;

#define BB 4
#define CC 256
#define HH 64
#define WW 64
#define HP (HH + 2)     // padded height (1-pixel zero halo)
#define WP (WW + 2)     // padded width
#define KK 2304         // 9 * 256

__device__ __forceinline__ f16x16 cat8(f16x8 lo, f16x8 hi) {
  return __builtin_shufflevector(lo, hi, 0,1,2,3,4,5,6,7,8,9,10,11,12,13,14,15);
}

// ---------------------------------------------------------------------------
// Branch-free implicit-GEMM mainloop over a zero-padded NHWC f16 image.
// One wave computes an MT*16(M) x NT*16(N) tile of
//   out[p][n] = sum_{tap,c} Apad[neighbor(p,tap)][c] * Bw[n][tap*256+c]
// A fragment (16-bit A 16x32): lane = m + 16*g; halves i<8 -> K=8g+i,
// i>=8 -> K=16+8g+(i-8).  B fragment (16-bit B 32x16): lane = n + 16*h;
// same K order as A: i<8 -> K=8h+i, i>=8 -> K=16+8h+(i-8).  (CDNA5 ISA 7.12.2)
// Per K-step: 2*MT A loads + 2*NT B loads feed MT*NT WMMAs.
// ---------------------------------------------------------------------------
template<int MT, int NT>
__device__ __forceinline__ void conv_mainloop(
    const _Float16* __restrict__ Apad, const _Float16* __restrict__ Bw,
    int m0, int n0, f32x8 acc[MT][NT]) {
  const int lane = threadIdx.x & 31;
  const int mrow = lane & 15;     // M row (A) / N col (B)
  const int gsel = lane >> 4;     // K-half select for both A and B
  const f32x8 zero8f = {0.f,0.f,0.f,0.f,0.f,0.f,0.f,0.f};

  const _Float16* abase[MT];
  #pragma unroll
  for (int mi = 0; mi < MT; ++mi) {
    int p = m0 + mi*16 + mrow;
    int b = p >> 12, h = (p >> 6) & 63, w = p & 63;
    abase[mi] = Apad + (((size_t)((b*HP + (h+1))*WP + (w+1))) << 8) + 8*gsel;
    #pragma unroll
    for (int j = 0; j < NT; ++j) acc[mi][j] = zero8f;
  }

  #pragma unroll
  for (int dy = -1; dy <= 1; ++dy) {
    #pragma unroll
    for (int dx = -1; dx <= 1; ++dx) {
      const int kk0 = ((dy + 1)*3 + (dx + 1)) << 8;        // tap * 256
      const int ashift = (dy*WP + dx) << 8;                // pixel shift * C
      #pragma unroll
      for (int c0 = 0; c0 < CC; c0 += 32) {
        f16x16 afrag[MT];
        #pragma unroll
        for (int mi = 0; mi < MT; ++mi) {
          const _Float16* ap = abase[mi] + ashift + c0;
          afrag[mi] = cat8(*(const f16x8*)ap, *(const f16x8*)(ap + 16));
        }
        #pragma unroll
        for (int j = 0; j < NT; ++j) {
          const _Float16* bp = Bw + (size_t)(n0 + j*16 + mrow)*KK
                                  + (kk0 + c0) + 8*gsel;
          f16x16 bfrag = cat8(*(const f16x8*)bp, *(const f16x8*)(bp + 16));
          #pragma unroll
          for (int mi = 0; mi < MT; ++mi)
            acc[mi][j] = __builtin_amdgcn_wmma_f32_16x16x32_f16(
                false, afrag[mi], false, bfrag, (short)0, acc[mi][j],
                false, false);
        }
      }
    }
  }
}

// --- zero fill (halo init for padded NHWC buffers) -------------------------
__global__ void def_zero_kernel(uint32_t* __restrict__ p, int n) {
  int i = blockIdx.x * 256 + threadIdx.x;
  if (i < n) p[i] = 0u;
}

// --- x: NCHW fp32 -> padded NHWC f16 (LDS tile transpose) ------------------
__global__ void def_pack_x_kernel(const float* __restrict__ x,
                                  _Float16* __restrict__ xh) {
  __shared__ float tile[16][17];
  int c0 = blockIdx.x * 16, w0 = blockIdx.y * 16;
  int bh = blockIdx.z;
  int b = bh >> 6, h = bh & 63;
  int tx = threadIdx.x, ty = threadIdx.y;
  tile[ty][tx] = x[(((size_t)b*CC + (c0 + ty))*HH + h)*WW + (w0 + tx)];
  __syncthreads();
  xh[(((size_t)b*HP + (h+1))*WP + (w0 + ty + 1))*CC + (c0 + tx)] =
      (_Float16)tile[tx][ty];
}

// --- weights [O,I,3,3] fp32 -> [O_pad][9*256] f16 (k = tap*256 + c) --------
__global__ void def_repack_w_kernel(const float* __restrict__ src,
                                    _Float16* __restrict__ dst,
                                    int o_real, int n_total) {
  int idx = blockIdx.x * 256 + threadIdx.x;
  if (idx >= n_total) return;
  int o = idx / KK, k = idx % KK;
  int t = k >> 8, c = k & 255;
  float v = (o < o_real) ? src[((size_t)o*CC + c)*9 + t] : 0.f;
  dst[idx] = (_Float16)v;
}

// --- offset conv: M=16384, N=32(pad, 18 real), K=2304 ----------------------
__global__ void __launch_bounds__(32)
def_offset_gemm_kernel(const _Float16* __restrict__ xh,
                       const _Float16* __restrict__ wofft,
                       const float* __restrict__ offset_b,
                       float* __restrict__ offbuf) {
  f32x8 acc[2][2];
  int m0 = blockIdx.x * 32;
  conv_mainloop<2, 2>(xh, wofft, m0, 0, acc);
  int lane = threadIdx.x & 31;
  int ncol = lane & 15, g = lane >> 4;
  #pragma unroll
  for (int mi = 0; mi < 2; ++mi) {
    #pragma unroll
    for (int j = 0; j < 2; ++j) {
      int n = j*16 + ncol;
      if (n < 18) {
        float bias = offset_b[n];
        #pragma unroll
        for (int r = 0; r < 8; ++r) {
          int p = m0 + mi*16 + 8*g + r;      // D layout: VGPR r -> M = r + 8*g
          offbuf[(size_t)p*18 + n] = acc[mi][j][r] + bias;
        }
      }
    }
  }
}

// --- 9-tap bilinear sample + mean; padded NHWC f16 in -> padded NHWC f16 ---
__global__ void def_sample_kernel(const _Float16* __restrict__ xh,
                                  const float* __restrict__ offb,
                                  _Float16* __restrict__ defh) {
  __shared__ int   sx[9], sy[9];
  __shared__ float swx[9], swy[9];
  int p = blockIdx.x;
  int b = p >> 12, h = (p >> 6) & 63, w = p & 63;
  int tid = threadIdx.x;
  if (tid < 9) {
    float ox = offb[(size_t)p*18 + 2*tid];
    float oy = offb[(size_t)p*18 + 2*tid + 1];
    // align_corners=True: unnormalize(normalize(pos)) == pos
    float ix = (float)w + ox;
    float iy = (float)h + oy;
    float fx = floorf(ix), fy = floorf(iy);
    sx[tid] = (int)fx;  sy[tid] = (int)fy;
    swx[tid] = ix - fx; swy[tid] = iy - fy;
  }
  __syncthreads();
  int c = tid;
  const size_t bbase = (size_t)b * (HP*WP*CC);
  float acc = 0.f;
  for (int t = 0; t < 9; ++t) {
    int x0 = sx[t], y0 = sy[t];
    float wx1 = swx[t], wy1 = swy[t];
    float wx0 = 1.f - wx1, wy0 = 1.f - wy1;
    float v00 = 0.f, v10 = 0.f, v01 = 0.f, v11 = 0.f;
    if (y0 >= 0 && y0 < HH) {
      size_t row = bbase + ((size_t)(y0+1) * WP) * CC;
      if (x0 >= 0   && x0   < WW) v00 = (float)xh[row + (size_t)(x0+1)*CC + c];
      if (x0+1 >= 0 && x0+1 < WW) v10 = (float)xh[row + (size_t)(x0+2)*CC + c];
    }
    if (y0+1 >= 0 && y0+1 < HH) {
      size_t row = bbase + ((size_t)(y0+2) * WP) * CC;
      if (x0 >= 0   && x0   < WW) v01 = (float)xh[row + (size_t)(x0+1)*CC + c];
      if (x0+1 >= 0 && x0+1 < WW) v11 = (float)xh[row + (size_t)(x0+2)*CC + c];
    }
    acc += wx0*wy0*v00 + wx1*wy0*v10 + wx0*wy1*v01 + wx1*wy1*v11;
  }
  defh[bbase + (((size_t)(h+1))*WP + (w+1))*CC + c] = (_Float16)(acc * (1.f/9.f));
}

// --- main conv: M=16384, N=256, K=2304; 64x64 tile per wave ----------------
__global__ void __launch_bounds__(32)
def_main_gemm_kernel(const _Float16* __restrict__ defh,
                     const _Float16* __restrict__ wt,
                     const float* __restrict__ conv_b,
                     float* __restrict__ ybuf) {
  f32x8 acc[4][4];
  int m0 = blockIdx.x * 64;
  int n0 = blockIdx.y * 64;
  conv_mainloop<4, 4>(defh, wt, m0, n0, acc);
  int lane = threadIdx.x & 31;
  int ncol = lane & 15, g = lane >> 4;
  #pragma unroll
  for (int mi = 0; mi < 4; ++mi) {
    #pragma unroll
    for (int j = 0; j < 4; ++j) {
      int n = n0 + j*16 + ncol;
      float bias = conv_b[n];
      #pragma unroll
      for (int r = 0; r < 8; ++r) {
        int p = m0 + mi*16 + 8*g + r;
        int b = p >> 12, h = (p >> 6) & 63, w = p & 63;
        ybuf[(((size_t)b*CC + n)*HH + h)*WW + w] = acc[mi][j][r] + bias;
      }
    }
  }
}

// --- per-channel sum / sumsq (one block per channel, no atomics) -----------
__global__ void def_stats_kernel(const float* __restrict__ y,
                                 float* __restrict__ stats) {
  __shared__ float ssum[256], ssq[256];
  int n = blockIdx.x, tid = threadIdx.x;
  float s = 0.f, q = 0.f;
  for (int b = 0; b < BB; ++b) {
    const float* plane = y + ((size_t)b*CC + n)*(HH*WW);
    for (int i = tid; i < HH*WW; i += 256) {
      float v = plane[i];
      s += v; q += v*v;
    }
  }
  ssum[tid] = s; ssq[tid] = q;
  __syncthreads();
  for (int off = 128; off > 0; off >>= 1) {
    if (tid < off) { ssum[tid] += ssum[tid+off]; ssq[tid] += ssq[tid+off]; }
    __syncthreads();
  }
  if (tid == 0) { stats[n] = ssum[0]; stats[CC + n] = ssq[0]; }
}

// --- batchnorm (biased var) + relu, NCHW fp32 ------------------------------
__global__ void def_bn_relu_kernel(const float* __restrict__ y,
                                   const float* __restrict__ stats,
                                   const float* __restrict__ gamma,
                                   const float* __restrict__ beta,
                                   float* __restrict__ out) {
  int idx = blockIdx.x * 256 + threadIdx.x;
  int n = (idx >> 12) & (CC - 1);
  const float invN = 1.f / (float)(BB*HH*WW);
  float mean = stats[n] * invN;
  float var  = stats[CC + n] * invN - mean*mean;
  float inv  = rsqrtf(var + 1e-5f);
  float v = (y[idx] - mean) * inv;
  float o = gamma[n]*v + beta[n];
  out[idx] = fmaxf(o, 0.f);
}

extern "C" void kernel_launch(void* const* d_in, const int* in_sizes, int n_in,
                              void* d_out, int out_size, void* d_ws, size_t ws_size,
                              hipStream_t stream) {
  const float* x        = (const float*)d_in[0];
  const float* offset_w = (const float*)d_in[1];
  const float* offset_b = (const float*)d_in[2];
  const float* conv_w   = (const float*)d_in[3];
  const float* conv_b   = (const float*)d_in[4];
  const float* gamma    = (const float*)d_in[5];
  const float* beta     = (const float*)d_in[6];
  float* out = (float*)d_out;

  // Workspace layout (256B-aligned chunks)
  char* ws = (char*)d_ws;
  size_t off = 0;
  auto alloc = [&](size_t bytes) {
    char* p = ws + off;
    off += (bytes + 255) & ~(size_t)255;
    return p;
  };
  const size_t padded_elems = (size_t)BB*HP*WP*CC;               // 4,460,544
  _Float16* xh    = (_Float16*)alloc(padded_elems * 2);          // padded NHWC f16
  _Float16* defh  = (_Float16*)alloc(padded_elems * 2);          // padded NHWC f16
  _Float16* wofft = (_Float16*)alloc((size_t)32*KK * 2);         // padded offset wts
  _Float16* wt    = (_Float16*)alloc((size_t)CC*KK * 2);         // main conv wts
  float*    offbf = (float*)   alloc((size_t)BB*HH*WW*18 * 4);   // offsets
  float*    ybuf  = (float*)   alloc((size_t)BB*CC*HH*WW * 4);   // pre-BN NCHW f32
  float*    stats = (float*)   alloc((size_t)2*CC * 4);          // sum / sumsq

  // 0) zero the padded buffers (halo must be zero for branch-free GEMM)
  const int zero_words = (int)(padded_elems * 2 / 4);
  def_zero_kernel<<<(zero_words + 255)/256, 256, 0, stream>>>((uint32_t*)xh,   zero_words);
  def_zero_kernel<<<(zero_words + 255)/256, 256, 0, stream>>>((uint32_t*)defh, zero_words);
  // 1) channels-last f16 image (interior of padded buffer)
  def_pack_x_kernel<<<dim3(CC/16, WW/16, BB*HH), dim3(16,16), 0, stream>>>(x, xh);
  // 2) weight repacks
  def_repack_w_kernel<<<(32*KK + 255)/256, 256, 0, stream>>>(offset_w, wofft, 18, 32*KK);
  def_repack_w_kernel<<<(CC*KK + 255)/256, 256, 0, stream>>>(conv_w, wt, CC, CC*KK);
  // 3) offset conv (WMMA implicit GEMM, 32x32 tile per wave)
  def_offset_gemm_kernel<<<dim3((BB*HH*WW)/32, 1), 32, 0, stream>>>(xh, wofft, offset_b, offbf);
  // 4) deformable 9-tap sampling + mean
  def_sample_kernel<<<BB*HH*WW, 256, 0, stream>>>(xh, offbf, defh);
  // 5) main conv (WMMA implicit GEMM, 64x64 tile per wave)
  def_main_gemm_kernel<<<dim3((BB*HH*WW)/64, CC/64), 32, 0, stream>>>(defh, wt, conv_b, ybuf);
  // 6) batch stats + 7) BN + ReLU
  def_stats_kernel<<<CC, 256, 0, stream>>>(ybuf, stats);
  def_bn_relu_kernel<<<(BB*CC*HH*WW)/256, 256, 0, stream>>>(ybuf, stats, gamma, beta, out);
}
